// BlockGCN_10_67473936221024
// MI455X (gfx1250) — matrix operand order repaired and checked by reading, not compiled, not run on hardware
//
#include <hip/hip_runtime.h>
#include <hip/hip_bf16.h>
#include <stdint.h>

// ---------------- problem constants ----------------
#define V     23
#define NK    3          // K kernels
#define H     8
#define C     128
#define OUTC  128
#define NB    32         // batch
#define T     256
#define CING  16
#define COUTG 16
#define RQ    368        // 16*23 (real rows/cols of folded matrix)
#define RQP   384        // padded to multiples of 16 / KSTEP
#define NTILE 32         // t columns per block
#define KSTEP 32         // bf16 WMMA K
#define NKT   (RQP / KSTEP)   // 12 k-iterations
#define BSTR  40         // LDS B row stride in bf16 halves (80B, 16B aligned)
#define OSTR  33         // LDS out-tile row stride in floats
#define BN_EPS 1e-5f

typedef __attribute__((ext_vector_type(16))) __bf16        v16bf;
typedef __attribute__((ext_vector_type(8)))  float         v8f;
typedef __attribute__((ext_vector_type(4)))  unsigned short us4;
typedef __attribute__((ext_vector_type(8)))  unsigned short us8;
typedef __attribute__((ext_vector_type(16))) unsigned short us16;

__device__ __forceinline__ unsigned short f2bf(float f) {
  unsigned u = __float_as_uint(f);
  u += 0x7FFFu + ((u >> 16) & 1u);      // round-to-nearest-even
  return (unsigned short)(u >> 16);
}

// ---------------------------------------------------------------------------
// Kernel 1: BnA[k,h,v,w] = B/l2col(B) + A/l2col(A), B = emb[k,h,hop[v,w]];
// last block folds bias + batchnorm into per-channel alpha/beta.
// ---------------------------------------------------------------------------
__global__ void prep_kernel(const float* __restrict__ emb,
                            const float* __restrict__ A,
                            const float* __restrict__ conv_b,
                            const float* __restrict__ bn_g,
                            const float* __restrict__ bn_b,
                            const float* __restrict__ bn_m,
                            const float* __restrict__ bn_v,
                            const int*   __restrict__ hop,
                            int n_hop,
                            float* __restrict__ BnA,
                            float* __restrict__ alpha,
                            float* __restrict__ beta) {
  int bid = blockIdx.x;
  if (bid == NK * H) {                       // per-channel epilogue constants
    int ch = threadIdx.x;
    if (ch < OUTC) {
      float inv = bn_g[ch] * rsqrtf(bn_v[ch] + BN_EPS);
      float bs = 0.f;
      for (int k = 0; k < NK; ++k) bs += conv_b[k * OUTC + ch];
      alpha[ch] = inv;
      beta[ch]  = bn_b[ch] - bn_m[ch] * inv + bs * inv;
    }
    return;
  }
  __shared__ float Bsh[V * V];
  __shared__ float nBc[V], nAc[V];
  const float* embkh = emb + (size_t)bid * n_hop;
  const float* Akh   = A   + (size_t)bid * V * V;
  for (int i = threadIdx.x; i < V * V; i += blockDim.x)
    Bsh[i] = embkh[hop[i]];
  __syncthreads();
  if (threadIdx.x < V) {                     // column norms (axis = v)
    int w = threadIdx.x;
    float sb = 0.f, sa = 0.f;
    for (int v = 0; v < V; ++v) {
      float b = Bsh[v * V + w]; sb += b * b;
      float a = Akh[v * V + w]; sa += a * a;
    }
    nBc[w] = sqrtf(sb) + 1e-4f;
    nAc[w] = sqrtf(sa) + 1e-4f;
  }
  __syncthreads();
  for (int i = threadIdx.x; i < V * V; i += blockDim.x) {
    int w = i % V;
    BnA[(size_t)bid * V * V + i] = Bsh[i] / nBc[w] + Akh[i] / nAc[w];
  }
}

// ---------------------------------------------------------------------------
// Kernel 2: fold conv weights and BnA (sum over k) into one bf16 matrix per
// head:  M_h[r=(o*23+w)][q=(c*23+v)] = sum_k W[k,h,o,c] * BnA[k,h,v,w]
// padded to 384x384 with zeros.  8 * 288KB bf16 -> L2 resident.
// ---------------------------------------------------------------------------
__global__ void buildM_kernel(const float* __restrict__ conv_w,
                              const float* __restrict__ BnA,
                              unsigned short* __restrict__ Mb) {
  int h = blockIdx.x;
  for (int idx = threadIdx.x; idx < RQP * RQP; idx += blockDim.x) {
    int r = idx / RQP, q = idx % RQP;
    float s = 0.f;
    if (r < RQ && q < RQ) {
      int o = r / V, w = r - o * V;
      int c = q / V, v = q - c * V;
#pragma unroll
      for (int k = 0; k < NK; ++k) {
        float wk = conv_w[(((k * H + h) * COUTG + o) * CING) + c];
        float bn = BnA[(((size_t)(k * H + h)) * V + v) * V + w];
        s += wk * bn;
      }
    }
    Mb[(size_t)h * RQP * RQP + idx] = f2bf(s);
  }
}

// ---------------------------------------------------------------------------
// Kernel 3: per (n, h, t-tile) block: out_tile(384x32) = M_h(384x384) * X(384x32)
// bf16 WMMA 16x16x32, fp32 accumulate; LDS-staged coalesced epilogue.
// ---------------------------------------------------------------------------
__global__ __launch_bounds__(256) void gemm_kernel(
    const float* __restrict__ x,
    const unsigned short* __restrict__ Mb,
    const float* __restrict__ alpha,
    const float* __restrict__ beta,
    float* __restrict__ out) {
  __shared__ unsigned short Bs[2][NTILE * BSTR];   // bf16 B tiles, double buffer
  __shared__ float Ot[RQ * OSTR];                  // staged output tile
  __shared__ float sAlpha[COUTG], sBeta[COUTG];

  const int t0   = blockIdx.x * NTILE;
  const int h    = blockIdx.y;
  const int n    = blockIdx.z;
  const int tid  = threadIdx.x;
  const int lane = tid & 31;
  const int wv   = tid >> 5;                       // wave 0..7
  const int l16  = lane & 15;
  const int hiHalf = lane >> 4;                    // 0/1: K half select

  if (tid < COUTG) {
    sAlpha[tid] = alpha[h * COUTG + tid];
    sBeta[tid]  = beta[h * COUTG + tid];
  }

  const float* xnh = x + (((size_t)n * C + h * CING) * T) * V;   // c=0 base
  const unsigned short* Mh = Mb + (size_t)h * RQP * RQP;

  v8f zero;
#pragma unroll
  for (int e = 0; e < 8; ++e) zero[e] = 0.f;
  v8f acc[3][2];
#pragma unroll
  for (int i = 0; i < 3; ++i) { acc[i][0] = zero; acc[i][1] = zero; }

  const int stq = (tid & 7) * 4;                   // q offset within K-chunk
  const int stt = tid >> 3;                        // t column 0..31

  // branch-free staging load: clamp address, select-zero after
  float ld[4];
#define STAGE_LOAD(KKBASE)                                                    \
  {                                                                           \
    _Pragma("unroll") for (int i_ = 0; i_ < 4; ++i_) {                        \
      int q_  = (KKBASE) + stq + i_;                                          \
      int qc_ = q_ < RQ ? q_ : 0;                                             \
      int c_ = qc_ / V, v_ = qc_ - c_ * V;                                    \
      float t_ = xnh[((size_t)c_ * T + (t0 + stt)) * V + v_];                 \
      ld[i_] = q_ < RQ ? t_ : 0.f;                                            \
    }                                                                         \
  }

#define STAGE_STORE(BUF)                                                      \
  {                                                                           \
    us4 st_;                                                                  \
    _Pragma("unroll") for (int i_ = 0; i_ < 4; ++i_) st_[i_] = f2bf(ld[i_]);  \
    *(us4*)&Bs[BUF][stt * BSTR + stq] = st_;                                  \
  }

  STAGE_LOAD(0)
  STAGE_STORE(0)

  for (int kt = 0; kt < NKT; ++kt) {
    const int kk = kt * KSTEP;
    const int p  = kt & 1;
    __syncthreads();                               // Bs[p] ready for all waves

    if (kt + 1 < NKT) STAGE_LOAD(kk + KSTEP)       // overlap next global loads

    if (kt + 1 < NKT)                              // near-scope prefetch of next A chunk
      __builtin_prefetch(Mh + (size_t)(wv * 48 + l16) * RQP + kk + KSTEP, 0, 3);

    // ---- B operands: lane = column t, K octets {0..7}/{8..15} then {16..23}/{24..31}
    v16bf bop[2];
#pragma unroll
    for (int ct = 0; ct < 2; ++ct) {
      const unsigned short* src = &Bs[p][(ct * 16 + l16) * BSTR + hiHalf * 8];
      us8 b0 = *(const us8*)(src);
      us8 b1 = *(const us8*)(src + 16);
      us16 bt = __builtin_shufflevector(b0, b1, 0, 1, 2, 3, 4, 5, 6, 7,
                                        8, 9, 10, 11, 12, 13, 14, 15);
      bop[ct] = __builtin_bit_cast(v16bf, bt);
    }

    // ---- load ALL three A operands first (one clause, overlapped waits) ----
    v16bf aop[3];
#pragma unroll
    for (int i = 0; i < 3; ++i) {
      int r = (wv * 3 + i) * 16 + l16;             // row of M_h
      const unsigned short* ga = Mh + (size_t)r * RQP + kk + hiHalf * 8;
      us8 a0 = *(const us8*)(ga);                  // K octet {0..7}/{8..15}
      us8 a1 = *(const us8*)(ga + 16);             // K octet {16..23}/{24..31}
      us16 at = __builtin_shufflevector(a0, a1, 0, 1, 2, 3, 4, 5, 6, 7,
                                        8, 9, 10, 11, 12, 13, 14, 15);
      aop[i] = __builtin_bit_cast(v16bf, at);
    }

    // ---- 6 WMMAs ----
#pragma unroll
    for (int i = 0; i < 3; ++i) {
      acc[i][0] = __builtin_amdgcn_wmma_f32_16x16x32_bf16(
          false, aop[i], false, bop[0], (short)0, acc[i][0], false, false);
      acc[i][1] = __builtin_amdgcn_wmma_f32_16x16x32_bf16(
          false, aop[i], false, bop[1], (short)0, acc[i][1], false, false);
    }

    if (kt + 1 < NKT) STAGE_STORE((kt + 1) & 1)
  }

  // ---- stage scaled tile to LDS (disjoint cells per lane, no barrier needed)
#pragma unroll
  for (int i = 0; i < 3; ++i) {
    int rt = wv * 3 + i;
#pragma unroll
    for (int j = 0; j < 8; ++j) {
      int r = rt * 16 + j + hiHalf * 8;            // C/D layout: vgpr j, M=j(+8)
      if (r < RQ) {
        int o = r / V;
        float a = sAlpha[o], b = sBeta[o];
#pragma unroll
        for (int ct = 0; ct < 2; ++ct) {
          int tl = ct * 16 + l16;
          Ot[r * OSTR + tl] = acc[i][ct][j] * a + b;
        }
      }
    }
  }
  __syncthreads();

  // ---- fully coalesced residual + ReLU + store:
  // for fixed (n,ch) the region [t0..t0+32) x V is one contiguous 736-float run
  const float* xres = x + (((size_t)n * C + h * CING) * T + t0) * V;
  float*       ores = out + (((size_t)n * C + h * CING) * T + t0) * V;
#pragma unroll
  for (int it = 0; it < (COUTG * NTILE * V) / 256; ++it) {   // 46 iterations
    int j = it * 256 + tid;                      // flat (o, tl, w)
    int o = j / (NTILE * V);
    int rem = j - o * (NTILE * V);
    int tl = rem / V;
    int w  = rem - tl * V;
    float val = Ot[(o * V + w) * OSTR + tl];
    size_t gidx = (size_t)o * T * V + rem;       // contiguous within o-segment
    val += xres[gidx];
    ores[gidx] = val > 0.f ? val : 0.f;
  }
}

// ---------------------------------------------------------------------------
extern "C" void kernel_launch(void* const* d_in, const int* in_sizes, int n_in,
                              void* d_out, int out_size, void* d_ws, size_t ws_size,
                              hipStream_t stream) {
  const float* x      = (const float*)d_in[0];
  const float* emb    = (const float*)d_in[1];
  const float* A      = (const float*)d_in[2];
  const float* conv_w = (const float*)d_in[3];
  const float* conv_b = (const float*)d_in[4];
  const float* bn_g   = (const float*)d_in[5];
  const float* bn_b   = (const float*)d_in[6];
  const float* bn_m   = (const float*)d_in[7];
  const float* bn_v   = (const float*)d_in[8];
  const int*   hop    = (const int*)d_in[9];
  const int n_hop = in_sizes[1] / (NK * H);

  // workspace layout: BnA (12696 f32) | alpha(128) | beta(128) | Mb (8*384*384 bf16)
  float* BnA   = (float*)d_ws;
  float* alpha = BnA + NK * H * V * V;
  float* beta  = alpha + OUTC;
  unsigned short* Mb = (unsigned short*)(beta + OUTC);

  prep_kernel<<<NK * H + 1, 256, 0, stream>>>(emb, A, conv_b, bn_g, bn_b, bn_m,
                                              bn_v, hop, n_hop, BnA, alpha, beta);
  buildM_kernel<<<H, 256, 0, stream>>>(conv_w, BnA, Mb);

  dim3 grid(T / NTILE, H, NB);
  gemm_kernel<<<grid, 256, 0, stream>>>(x, Mb, alpha, beta, (float*)d_out);
}
